// Transformer_75265006895297
// MI455X (gfx1250) — matrix operand order repaired and checked by reading, not compiled, not run on hardware
//
#include <hip/hip_runtime.h>
#include <hip/hip_bf16.h>
#include <math.h>

#define DIM   1024
#define DEPTH 4
#define VOCAB 32000
#define BB    2
#define NN    2048
#define TOK   (BB * NN)   // 4096 tokens
#define DFF   4096

#define TM 128
#define TN 128
#define TKS 32

typedef __attribute__((ext_vector_type(16))) __bf16 bf16x16;
typedef __attribute__((ext_vector_type(8)))  float  f32x8;
typedef __attribute__((ext_vector_type(4)))  int    i32x4;

#if defined(__has_builtin)
#if __has_builtin(__builtin_amdgcn_global_load_async_to_lds_b128) && \
    __has_builtin(__builtin_amdgcn_s_wait_asynccnt)
#define ASYNC_LDS 1
#endif
#endif

#ifdef ASYNC_LDS
typedef __attribute__((address_space(1))) i32x4 gas_i32x4;
typedef __attribute__((address_space(3))) i32x4 las_i32x4;
#endif

// Copy 16 bytes global -> LDS. Async variant uses the gfx1250
// GLOBAL_LOAD_ASYNC_TO_LDS_B128 path (tracked by ASYNCcnt).
__device__ __forceinline__ void cp16_g2l(const float* g, float* l) {
#ifdef ASYNC_LDS
  // generic->flat integer cast: global generic addr == as(1) addr; low 32
  // bits of a generic LDS addr == LDS byte offset (as(3) is 32-bit).
  __builtin_amdgcn_global_load_async_to_lds_b128(
      (gas_i32x4*)(unsigned long long)g,
      (las_i32x4*)(unsigned int)(unsigned long long)l,
      0, 0);
#else
  *(float4*)l = *(const float4*)g;
#endif
}

__device__ __forceinline__ void wait_stage() {
#ifdef ASYNC_LDS
  __builtin_amdgcn_s_wait_asynccnt(0);
#endif
  __syncthreads();
}

__device__ __forceinline__ float gelu_exact(float v) {
  return 0.5f * v * (1.0f + erff(v * 0.70710678118654752f));
}

// ------------------------------------------------------------------
// Embedding gather: one block per token, 256 threads x float4 = 1024 f32
// ------------------------------------------------------------------
__global__ __launch_bounds__(256) void embed_kernel(const int* __restrict__ tok,
                                                    const float* __restrict__ emb,
                                                    float* __restrict__ x) {
  const int t  = blockIdx.x;
  const int id = tok[t];
  const float4* src = (const float4*)(emb + (size_t)id * DIM);
  float4*       dst = (float4*)(x + (size_t)t * DIM);
  dst[threadIdx.x] = src[threadIdx.x];
}

// ------------------------------------------------------------------
// RMS norm: y = x / max(||x||, 1e-12) * sqrt(DIM) * g
// ------------------------------------------------------------------
__global__ __launch_bounds__(256) void rms_kernel(const float* __restrict__ x,
                                                  const float* __restrict__ g,
                                                  float* __restrict__ y) {
  __shared__ float red[256];
  const int t = blockIdx.x, tid = threadIdx.x;
  const float* xr = x + (size_t)t * DIM;
  float s = 0.f;
#pragma unroll
  for (int i = 0; i < DIM / 256; ++i) {
    float v = xr[tid + i * 256];
    s = fmaf(v, v, s);
  }
  red[tid] = s;
  __syncthreads();
  for (int o = 128; o > 0; o >>= 1) {
    if (tid < o) red[tid] += red[tid + o];
    __syncthreads();
  }
  const float scale = 32.0f / fmaxf(sqrtf(red[0]), 1e-12f);
  float* yr = y + (size_t)t * DIM;
#pragma unroll
  for (int i = 0; i < DIM / 256; ++i) {
    const int idx = tid + i * 256;
    yr[idx] = xr[idx] * scale * g[idx];
  }
}

// ------------------------------------------------------------------
// GEMM C[M,N] = A[M,K] (row-major f32) @ B[K,N] (row-major f32)
// bf16 WMMA 16x16x32. Block tile 128x128, 8 waves (256 threads),
// each wave owns a 32x64 subtile = 2x4 WMMA fragments.
// Double-buffered LDS; staging via async-to-LDS when available.
// ep: 0 = none, 1 = +bias, 2 = +bias then exact GELU
// ------------------------------------------------------------------
__global__ __launch_bounds__(256) void gemm_bf16_wmma(
    const float* __restrict__ A, const float* __restrict__ Bw,
    const float* __restrict__ bias, float* __restrict__ C,
    int M, int N, int K, int ep) {
  __shared__ __align__(16) float As[2][TM][TKS];  // 2 x 16 KB
  __shared__ __align__(16) float Bs[2][TKS][TN];  // 2 x 16 KB

  const int tid  = threadIdx.x;
  const int wave = tid >> 5, lane = tid & 31;
  const int hi = lane >> 4, lm = lane & 15;
  const int bm = blockIdx.y * TM, bn = blockIdx.x * TN;
  const int wm = (wave >> 1) << 5;  // 0,32,64,96
  const int wn = (wave & 1) << 6;   // 0,64

  f32x8 acc[2][4] = {};

  // staging assignments (16 floats per thread per tile, 4 x b128)
  const int ar_ = tid >> 1, ac_ = (tid & 1) << 4;  // A: 128 rows x 32 cols
  const int br_ = tid >> 3, bc_ = (tid & 7) << 4;  // B: 32 rows x 128 cols

  auto stage = [&](int k0, int buf) {
    const float* ga = A + (size_t)(bm + ar_) * K + (k0 + ac_);
    const float* gb = Bw + (size_t)(k0 + br_) * N + (bn + bc_);
    float* la = &As[buf][ar_][ac_];
    float* lb = &Bs[buf][br_][bc_];
#pragma unroll
    for (int i = 0; i < 4; ++i) cp16_g2l(ga + 4 * i, la + 4 * i);
#pragma unroll
    for (int i = 0; i < 4; ++i) cp16_g2l(gb + 4 * i, lb + 4 * i);
    if (k0 + TKS < K) {  // global_prefetch_b8 two tiles ahead of compute
      __builtin_prefetch(ga + TKS, 0, 1);
      __builtin_prefetch(gb + (size_t)TKS * N, 0, 1);
    }
  };

  stage(0, 0);
  wait_stage();

  int buf = 0;
  for (int k0 = 0; k0 < K; k0 += TKS) {
    const int knext = k0 + TKS;
    if (knext < K) stage(knext, buf ^ 1);

    // Build bf16 fragments per ISA 7.12.2 layouts.
    // A (16x32, 16-bit): lane = hi*16 + m; VGPR r half j:
    //   k = (r/4)*16 + hi*8 + (r%4)*2 + j
    bf16x16 af[2];
#pragma unroll
    for (int mi = 0; mi < 2; ++mi) {
      const float* rowp = As[buf][wm + (mi << 4) + lm];
#pragma unroll
      for (int r = 0; r < 8; ++r) {
        const int kb = ((r >> 2) << 4) + (hi << 3) + ((r & 3) << 1);
        af[mi][2 * r]     = (__bf16)rowp[kb];
        af[mi][2 * r + 1] = (__bf16)rowp[kb + 1];
      }
    }
    // B (32x16, 16-bit): same k order as A: k = (r/4)*16 + hi*8 + (r%4)*2 + j
    bf16x16 bfr[4];
#pragma unroll
    for (int ni = 0; ni < 4; ++ni) {
      const int col = wn + (ni << 4) + lm;
#pragma unroll
      for (int r = 0; r < 8; ++r) {
        const int kb = ((r >> 2) << 4) + (hi << 3) + ((r & 3) << 1);
        bfr[ni][2 * r]     = (__bf16)Bs[buf][kb][col];
        bfr[ni][2 * r + 1] = (__bf16)Bs[buf][kb + 1][col];
      }
    }
#pragma unroll
    for (int mi = 0; mi < 2; ++mi)
#pragma unroll
      for (int ni = 0; ni < 4; ++ni)
        acc[mi][ni] = __builtin_amdgcn_wmma_f32_16x16x32_bf16(
            false, af[mi], false, bfr[ni], (short)0, acc[mi][ni], false, false);

    wait_stage();  // drain ASYNCcnt for next buffer + block barrier
    buf ^= 1;
  }

  // Epilogue + store. D layout: VGPR r -> row = base + hi*8 + r, col = lane%16
#pragma unroll
  for (int mi = 0; mi < 2; ++mi) {
#pragma unroll
    for (int ni = 0; ni < 4; ++ni) {
      const int col  = bn + wn + (ni << 4) + lm;
      const int row0 = bm + wm + (mi << 4) + (hi << 3);
      const float bv = (ep != 0 && bias != nullptr) ? bias[col] : 0.f;
#pragma unroll
      for (int r = 0; r < 8; ++r) {
        float v = acc[mi][ni][r] + bv;
        if (ep == 2) v = gelu_exact(v);
        C[(size_t)(row0 + r) * N + col] = v;
      }
    }
  }
#if __has_builtin(__builtin_amdgcn_s_wait_tensorcnt)
  __builtin_amdgcn_s_wait_tensorcnt(0);
#endif
}

// ------------------------------------------------------------------
// GateLoop recurrence (head_dim==1 -> scalar per channel):
//   g[n] = sigmoid(|a|) * cos(angle(a)) = sigmoid(r) * ar / r
//   kv[n] = g[n]*kv[n-1] + k[n]*v[n];   out[n] = q[n]*kv[n]
// One wave32 per channel; lane handles 64 consecutive steps; carry via
// an inclusive __shfl_up scan of (G, U) pairs, then replay.
// ------------------------------------------------------------------
__device__ __forceinline__ float decay_g(float ar, float ai) {
  const float r = sqrtf(fmaf(ar, ar, ai * ai));
  const float sg = 1.0f / (1.0f + expf(-r));
  return (r > 0.f) ? sg * (ar / r) : 0.5f;
}

__global__ __launch_bounds__(128) void gate_scan_kernel(const float* __restrict__ qkv,
                                                        const float* __restrict__ ab,
                                                        float* __restrict__ out) {
  const int wave = threadIdx.x >> 5, lane = threadIdx.x & 31;
  const int ch = (blockIdx.x << 2) + wave;   // 0 .. B*DIM-1
  const int b = ch >> 10, d = ch & (DIM - 1);
  const int seg = NN / 32;                    // 64 steps per lane
  const size_t t0 = (size_t)b * NN + (size_t)lane * seg;

  // pass 1: per-lane segment reduction (G = prod g, U = segment state from 0)
  float G = 1.f, U = 0.f;
  for (int s = 0; s < seg; ++s) {
    const size_t t = t0 + s;
    const float kk = qkv[t * (3 * DIM) + DIM + d];
    const float vv = qkv[t * (3 * DIM) + 2 * DIM + d];
    const float ar = ab[t * (2 * DIM) + 2 * d];
    const float ai = ab[t * (2 * DIM) + 2 * d + 1];
    const float g = decay_g(ar, ai);
    U = fmaf(g, U, kk * vv);
    G *= g;
  }
  // inclusive wave scan, combine: (self) after (left) => U = G*U_left + U
  for (int off = 1; off < 32; off <<= 1) {
    const float pg = __shfl_up(G, off, 32);
    const float pu = __shfl_up(U, off, 32);
    if (lane >= off) { U = fmaf(G, pu, U); G *= pg; }
  }
  float kv = __shfl_up(U, 1, 32);  // exclusive carry
  if (lane == 0) kv = 0.f;

  // pass 2: replay with carry, emit q*kv
  for (int s = 0; s < seg; ++s) {
    const size_t t = t0 + s;
    const float qq = qkv[t * (3 * DIM) + d];
    const float kk = qkv[t * (3 * DIM) + DIM + d];
    const float vv = qkv[t * (3 * DIM) + 2 * DIM + d];
    const float ar = ab[t * (2 * DIM) + 2 * d];
    const float ai = ab[t * (2 * DIM) + 2 * d + 1];
    const float g = decay_g(ar, ai);
    kv = fmaf(g, kv, kk * vv);
    out[t * DIM + d] = qq * kv;
  }
}

// ------------------------------------------------------------------
// Elementwise
// ------------------------------------------------------------------
__global__ __launch_bounds__(256) void silu_gate_kernel(const float* __restrict__ g,
                                                        float* __restrict__ o, int n) {
  const int i = blockIdx.x * blockDim.x + threadIdx.x;
  if (i < n) {
    const float gv = g[i];
    o[i] = o[i] * (gv / (1.0f + expf(-gv)));
  }
}

__global__ __launch_bounds__(256) void add_kernel(float* __restrict__ x,
                                                  const float* __restrict__ y, int n) {
  const int i = blockIdx.x * blockDim.x + threadIdx.x;
  if (i < n) x[i] += y[i];
}

// ------------------------------------------------------------------
// Orchestration
// ------------------------------------------------------------------
extern "C" void kernel_launch(void* const* d_in, const int* in_sizes, int n_in,
                              void* d_out, int out_size, void* d_ws, size_t ws_size,
                              hipStream_t stream) {
  (void)in_sizes; (void)n_in; (void)out_size; (void)ws_size;
  const int*   tokens = (const int*)  d_in[0];
  const float* emb    = (const float*)d_in[1];
  const float* g1     = (const float*)d_in[2];
  const float* Wqkv   = (const float*)d_in[3];
  const float* Wa     = (const float*)d_in[4];
  const float* ba     = (const float*)d_in[5];
  const float* Wg     = (const float*)d_in[6];
  const float* Wo     = (const float*)d_in[7];
  const float* g2     = (const float*)d_in[8];
  const float* W1     = (const float*)d_in[9];
  const float* b1     = (const float*)d_in[10];
  const float* W2     = (const float*)d_in[11];
  const float* b2     = (const float*)d_in[12];
  const float* gf     = (const float*)d_in[13];
  const float* Wl     = (const float*)d_in[14];
  float* logits = (float*)d_out;

  float* ws = (float*)d_ws;
  float* X  = ws;                              // [TOK, DIM]
  float* XN = X  + (size_t)TOK * DIM;          // [TOK, DIM]
  float* T3 = XN + (size_t)TOK * DIM;          // [TOK, max(3*DIM, DFF)]
  float* TA = T3 + (size_t)TOK * DFF;          // [TOK, 2*DIM]
  float* TG = TA + (size_t)TOK * 2 * DIM;      // [TOK, DIM]
  float* TO = TG + (size_t)TOK * DIM;          // [TOK, DIM]
  float* T2 = TO + (size_t)TOK * DIM;          // [TOK, DIM]

  auto gemm = [&](const float* Aa, const float* Bb, const float* bias, float* Cc,
                  int M, int Nn, int Kk, int ep) {
    dim3 grid(Nn / TN, M / TM);
    gemm_bf16_wmma<<<grid, dim3(256), 0, stream>>>(Aa, Bb, bias, Cc, M, Nn, Kk, ep);
  };

  embed_kernel<<<TOK, 256, 0, stream>>>(tokens, emb, X);

  const int nelem = TOK * DIM;
  for (int i = 0; i < DEPTH; ++i) {
    rms_kernel<<<TOK, 256, 0, stream>>>(X, g1 + (size_t)i * DIM, XN);
    gemm(XN, Wqkv + (size_t)i * DIM * 3 * DIM, nullptr, T3, TOK, 3 * DIM, DIM, 0);
    gemm(XN, Wa + (size_t)i * DIM * 2 * DIM, ba + (size_t)i * 2 * DIM, TA, TOK, 2 * DIM, DIM, 1);
    gemm(XN, Wg + (size_t)i * DIM * DIM, nullptr, TG, TOK, DIM, DIM, 0);
    gate_scan_kernel<<<(BB * DIM) / 4, 128, 0, stream>>>(T3, TA, TO);
    silu_gate_kernel<<<(nelem + 255) / 256, 256, 0, stream>>>(TG, TO, nelem);
    gemm(TO, Wo + (size_t)i * DIM * DIM, nullptr, T2, TOK, DIM, DIM, 0);
    add_kernel<<<(nelem + 255) / 256, 256, 0, stream>>>(X, T2, nelem);

    rms_kernel<<<TOK, 256, 0, stream>>>(X, g2 + (size_t)i * DIM, XN);
    gemm(XN, W1 + (size_t)i * DIM * DFF, b1 + (size_t)i * DFF, T3, TOK, DFF, DIM, 2);
    gemm(T3, W2 + (size_t)i * DFF * DIM, b2 + (size_t)i * DIM, T2, TOK, DIM, DFF, 1);
    add_kernel<<<(nelem + 255) / 256, 256, 0, stream>>>(X, T2, nelem);
  }

  rms_kernel<<<TOK, 256, 0, stream>>>(X, gf, XN);
  gemm(XN, Wl, nullptr, logits, TOK, VOCAB, DIM, 0);
}
